// SparseSubdivideBlock3d_60206851555800
// MI455X (gfx1250) — compile-verified, matrix operand order checked by reading, not run on hardware
//
#include <hip/hip_runtime.h>
#include <math.h>

#define N1   40000   // active voxels, layer 1
#define N2   320000  // 8 * N1 children, layer 2
#define C    128     // channels (in == out)
#define KOFF 27      // 3x3x3 offsets

typedef __attribute__((ext_vector_type(16))) __bf16 v16bf;
typedef __attribute__((ext_vector_type(8)))  __bf16 v8bf;
typedef __attribute__((ext_vector_type(8)))  float  v8f;

// ---------------------------------------------------------------------------
// Prep: convert feats f32 -> bf16, append one zero pad row (row N1).
// ---------------------------------------------------------------------------
__global__ void prep_feats_kernel(const float* __restrict__ feats,
                                  __bf16* __restrict__ featsA) {
    int i = blockIdx.x * blockDim.x + threadIdx.x;
    const int total = (N1 + 1) * C;
    if (i >= total) return;
    featsA[i] = (i < N1 * C) ? (__bf16)feats[i] : (__bf16)0.0f;
}

__global__ void zero_pad_kernel(__bf16* __restrict__ h1) {
    h1[(size_t)N1 * C + threadIdx.x] = (__bf16)0.0f;
}

// ---------------------------------------------------------------------------
// Repack W[k][cin][cout] (f32) into per-lane WMMA B-fragment layout (bf16).
// Block index = ((k*8 + nt)*4 + kk), block = 32 lanes x 16 bf16.
// Lane L -> col N = nt*16 + (L&15), khalf = L>>4; element e maps to
// kloc = (e<8 ? e : e+8) + khalf*8  (ISA 7.12.2 16-bit operand layout).
// ---------------------------------------------------------------------------
__global__ void repack_w_kernel(const float* __restrict__ W,
                                __bf16* __restrict__ Wp) {
    int i = blockIdx.x * blockDim.x + threadIdx.x;
    const int TOT = KOFF * 8 * 4 * 32 * 16;   // 442368
    if (i >= TOT) return;
    int e    = i & 15;
    int lane = (i >> 4) & 31;
    int kk   = (i >> 9) & 3;
    int nt   = (i >> 11) & 7;
    int k    = i >> 14;
    int n     = nt * 16 + (lane & 15);
    int khalf = lane >> 4;
    int kloc  = (e < 8 ? e : e + 8) + khalf * 8;
    int cin   = kk * 32 + kloc;
    Wp[i] = (__bf16)W[((size_t)k * C + cin) * C + n];
}

// ---------------------------------------------------------------------------
// Fused sparse-conv GEMM + bias + SiLU.
// MODE 0: layer 1 (gather row = nbr value, bf16 out to h1 table)
// MODE 1: layer 2 (gather row = nbr >> 3 == parent, f32 out)
// Block: 128 threads = 4 waves; wave w owns output cols [32w, 32w+32)
// (two 16-col WMMA N-tiles). Block tile = 64 rows x 128 cols.
// Rulebook tile (64 x 27 ints) staged in LDS once per block.
// k-loop kept rolled (32 WMMA + 48 b128 loads per body) so the hot loop
// stays I$-resident instead of 44KB of straight-line code.
// ---------------------------------------------------------------------------
template <int MODE>
__global__ __launch_bounds__(128)
void sparse_gemm_kernel(const __bf16* __restrict__ fin,   // (rows+1) x 128 bf16, last row zero
                        const __bf16* __restrict__ Wp,    // repacked weights
                        const float*  __restrict__ bias,  // 128
                        const int*    __restrict__ nbr,   // Mtot x 27
                        __bf16* __restrict__ outb,        // MODE 0 dest
                        float*  __restrict__ outf)        // MODE 1 dest
{
    __shared__ int snbr[64 * KOFF];   // 6912 B

    const int tid  = threadIdx.x;
    const int lane = tid & 31;
    const int w    = tid >> 5;                  // wave id -> col group (0..3)
    const int m0   = blockIdx.x * 64;           // first output row of this block
    const int r    = lane & 15;                 // M row within subtile
    const unsigned abase = (lane & 16) ? 16u : 0u;  // A-frag K sub-offset (bytes)

    // cooperative, coalesced rulebook stage (with parent mapping pre-applied)
    const int* gnb = nbr + (size_t)m0 * KOFF;
    for (int i = tid; i < 64 * KOFF; i += 128) {
        int v = gnb[i];
        snbr[i] = MODE ? (v >> 3) : v;
    }
    __syncthreads();

    v8f acc[4][2] = {};
    const char* finb = (const char*)fin;

#pragma unroll 1
    for (int k = 0; k < KOFF; ++k) {
        // per-lane gathered rows for the 4 M-subtiles (LDS, conflict-free)
        unsigned a0 = (unsigned)snbr[( 0 + r) * KOFF + k] * (C * 2u);
        unsigned a1 = (unsigned)snbr[(16 + r) * KOFF + k] * (C * 2u);
        unsigned a2 = (unsigned)snbr[(32 + r) * KOFF + k] * (C * 2u);
        unsigned a3 = (unsigned)snbr[(48 + r) * KOFF + k] * (C * 2u);

        const __bf16* wb0 = Wp + ((size_t)(k * 8 + 2 * w + 0) * 4) * 512 + lane * 16;
        const __bf16* wb1 = Wp + ((size_t)(k * 8 + 2 * w + 1) * 4) * 512 + lane * 16;

#pragma unroll
        for (int kk = 0; kk < 4; ++kk) {
            // two B fragments (pre-repacked, contiguous 32B per lane)
            v16bf bf0 = *(const v16bf*)(wb0 + kk * 512);
            v16bf bf1 = *(const v16bf*)(wb1 + kk * 512);

            const unsigned cb = kk * 64u + abase;   // byte offset within row
            union AF { v16bf v; v8bf h[2]; } af;

#pragma unroll
            for (int s = 0; s < 4; ++s) {
                unsigned ro = (s == 0) ? a0 : (s == 1) ? a1 : (s == 2) ? a2 : a3;
                af.h[0] = *(const v8bf*)(finb + ro + cb);
                af.h[1] = *(const v8bf*)(finb + ro + cb + 32u);
                acc[s][0] = __builtin_amdgcn_wmma_f32_16x16x32_bf16(
                                false, af.v, false, bf0, (short)0, acc[s][0], false, false);
                acc[s][1] = __builtin_amdgcn_wmma_f32_16x16x32_bf16(
                                false, af.v, false, bf1, (short)0, acc[s][1], false, false);
            }
        }
    }

    // Epilogue: bias + SiLU.  C/D layout: VGPR j -> M = j + (lane>=16?8:0), N = lane&15
    const int   c0   = w * 32 + (lane & 15);
    const float bv0  = bias[c0];
    const float bv1  = bias[c0 + 16];
    const int   rofs = (lane & 16) ? 8 : 0;

#pragma unroll
    for (int s = 0; s < 4; ++s) {
        const int rowb = m0 + s * 16 + rofs;
#pragma unroll
        for (int j = 0; j < 8; ++j) {
            float x0 = acc[s][0][j] + bv0;
            float x1 = acc[s][1][j] + bv1;
            float y0 = x0 * (1.0f / (1.0f + __expf(-x0)));
            float y1 = x1 * (1.0f / (1.0f + __expf(-x1)));
            size_t o = (size_t)(rowb + j) * C + c0;
            if (MODE) {
                outf[o]      = y0;
                outf[o + 16] = y1;
            } else {
                outb[o]      = (__bf16)y0;
                outb[o + 16] = (__bf16)y1;
            }
        }
    }
}

// ---------------------------------------------------------------------------
// Launch
// ---------------------------------------------------------------------------
extern "C" void kernel_launch(void* const* d_in, const int* in_sizes, int n_in,
                              void* d_out, int out_size, void* d_ws, size_t ws_size,
                              hipStream_t stream) {
    (void)in_sizes; (void)n_in; (void)out_size; (void)ws_size;

    const float* feats = (const float*)d_in[0];
    const float* W1    = (const float*)d_in[1];
    const float* b1    = (const float*)d_in[2];
    const float* W2    = (const float*)d_in[3];
    const float* b2    = (const float*)d_in[4];
    const int*   nbr1  = (const int*)d_in[5];
    const int*   nbr2  = (const int*)d_in[6];
    float*       out   = (float*)d_out;

    // workspace layout (all sizes 256B-aligned)
    const size_t SZF = (size_t)(N1 + 1) * C * 2;      // 10,240,256 B
    const size_t SZW = (size_t)KOFF * C * C * 2;      //    884,736 B
    char* ws = (char*)d_ws;
    __bf16* featsA = (__bf16*)(ws);
    __bf16* h1     = (__bf16*)(ws + SZF);
    __bf16* W1p    = (__bf16*)(ws + 2 * SZF);
    __bf16* W2p    = (__bf16*)(ws + 2 * SZF + SZW);

    prep_feats_kernel<<<((N1 + 1) * C + 255) / 256, 256, 0, stream>>>(feats, featsA);
    zero_pad_kernel<<<1, 128, 0, stream>>>(h1);

    const int WTOT = KOFF * 8 * 4 * 32 * 16;
    repack_w_kernel<<<(WTOT + 255) / 256, 256, 0, stream>>>(W1, W1p);
    repack_w_kernel<<<(WTOT + 255) / 256, 256, 0, stream>>>(W2, W2p);

    // Layer 1: 40000 rows -> 625 blocks of 64 rows
    sparse_gemm_kernel<0><<<N1 / 64, 128, 0, stream>>>(featsA, W1p, b1, nbr1, h1, nullptr);
    // Layer 2: 320000 child rows -> 5000 blocks; input rows = parent (idx>>3)
    sparse_gemm_kernel<1><<<N2 / 64, 128, 0, stream>>>(h1, W2p, b2, nbr2, nullptr, out);
}
